// GRNAspects_15307263443312
// MI455X (gfx1250) — compile-verified, matrix operand order checked by reading, not run on hardware
//
#include <hip/hip_runtime.h>

// Problem constants (fixed by the reference)
#define FDIM 128         // F_IN == F_OUT
#define KNB  16          // neighbors per node
#define BSTR 136         // bf16 LDS row stride (elements): 272 B, 16B-aligned, bank-skewed

typedef __attribute__((ext_vector_type(16))) __bf16 v16bf;
typedef __attribute__((ext_vector_type(8)))  float  v8f;

union FragAB {
    v16bf v;
    unsigned u[8];
};
union FragC {
    v8f v;
    float f[8];
};

__device__ __forceinline__ unsigned f32_to_bf16_bits(float f) {
    union { float f; unsigned u; } cv; cv.f = f;
    unsigned u = cv.u;
    unsigned r = u + 0x7FFFu + ((u >> 16) & 1u);   // round-to-nearest-even
    return r >> 16;
}

__device__ __forceinline__ unsigned pack_bf16x2(float lo, float hi) {
    return (f32_to_bf16_bits(hi) << 16) | (f32_to_bf16_bits(lo) & 0xFFFFu);
}

// ---------------------------------------------------------------------------
// CDNA5 async copy: global -> LDS, 16 bytes per lane, tracked by ASYNCcnt.
//   LDS destination from lds_off; global source at gbase plus per-lane goff
// ---------------------------------------------------------------------------
__device__ __forceinline__ void async_copy_b128(unsigned lds_off,
                                                const void* gbase,
                                                unsigned goff) {
    asm volatile("global_load_async_to_lds_b128 %0, %1, %2"
                 :
                 : "v"(lds_off), "v"(goff), "s"(gbase)
                 : "memory");
}

__device__ __forceinline__ void wait_async_all() {
    asm volatile("s_wait_asynccnt 0x0" ::: "memory");
}

// ---------------------------------------------------------------------------
// Fragment gather per CDNA5 ISA 7.12.2 (wave32), bf16 16x16x32:
//  A: lane L -> row m = L&15, hi = L>>4; slots 0..7 = K kb..kb+7,
//     slots 8..15 = K kb+16..kb+23, kb = kchunk*32 + 8*hi.
//  B: identical pattern on columns. Both runs are 16 contiguous bytes -> b128.
// ---------------------------------------------------------------------------
__device__ __forceinline__ FragAB frag_from_u16(const unsigned short* row, int kb) {
    FragAB f;
    const uint4 r0 = *(const uint4*)(row + kb);
    const uint4 r1 = *(const uint4*)(row + kb + 16);
    f.u[0] = r0.x; f.u[1] = r0.y; f.u[2] = r0.z; f.u[3] = r0.w;
    f.u[4] = r1.x; f.u[5] = r1.y; f.u[6] = r1.z; f.u[7] = r1.w;
    return f;
}

// ---------------------------------------------------------------------------
// Kernel 1: fold the two linear layers (tiny, runs once):
//   B1 = Wa[:, :128] @ W ; B2 = Wa[:, 128:] @ W ; Wbf = bf16(W)
// ---------------------------------------------------------------------------
__global__ __launch_bounds__(FDIM)
void prep_weights_kernel(const float* __restrict__ W,
                         const float* __restrict__ Wa,
                         unsigned short* __restrict__ B1bf,
                         unsigned short* __restrict__ B2bf,
                         unsigned short* __restrict__ Wbf) {
    __shared__ float wa[2 * FDIM];
    const int o = blockIdx.x;
    const int j = threadIdx.x;
    for (int t = threadIdx.x; t < 2 * FDIM; t += blockDim.x)
        wa[t] = Wa[o * 2 * FDIM + t];
    __syncthreads();
    float b1 = 0.f, b2 = 0.f;
    for (int f = 0; f < FDIM; ++f) {
        const float w = W[f * FDIM + j];       // coalesced across j
        b1 += wa[f] * w;
        b2 += wa[FDIM + f] * w;
    }
    B1bf[o * FDIM + j] = (unsigned short)f32_to_bf16_bits(b1);
    B2bf[o * FDIM + j] = (unsigned short)f32_to_bf16_bits(b2);
    Wbf[o * FDIM + j]  = (unsigned short)f32_to_bf16_bits(W[o * FDIM + j]);
}

// ---------------------------------------------------------------------------
// Kernel 2: node_proj[n, o] = sum_f nodes[n,f] * W[o,f]  (bf16 WMMA)
// Block = 256 thr = 8 waves; covers 16 nodes; wave w -> 16 output cols.
// Two interleaved accumulator chains (even/odd K-chunks) to dodge the
// bf16 WMMA->WMMA hazard NOPs.
// ---------------------------------------------------------------------------
__global__ __launch_bounds__(256)
void node_proj_kernel(const float* __restrict__ nodes,
                      const unsigned short* __restrict__ Wbf,
                      float* __restrict__ node_proj) {
    __shared__ __align__(16) float           stage[16 * FDIM];   // 8 KB fp32
    __shared__ __align__(16) unsigned short  abf[16][BSTR];      // bf16 tile

    const int tid = threadIdx.x;
    const float* gsrc = nodes + (size_t)blockIdx.x * 16 * FDIM;
    const unsigned sbase = (unsigned)(uintptr_t)&stage[0];

#pragma unroll
    for (int i = 0; i < 2; ++i) {
        const unsigned off = (unsigned)(tid + i * 256) * 16u;    // 512 x b128
        async_copy_b128(sbase + off, gsrc, off);
    }
    wait_async_all();
    __syncthreads();

    // fp32 -> bf16 (padded) : each thread converts 8 floats
#pragma unroll
    for (int i = 0; i < 2; ++i) {
        const int t  = tid + i * 256;       // float4 index 0..511
        const int r  = t >> 5;              // 32 float4 per 128-float row
        const int c4 = t & 31;
        const float4 v = ((const float4*)stage)[t];
        uint2 p; p.x = pack_bf16x2(v.x, v.y); p.y = pack_bf16x2(v.z, v.w);
        *(uint2*)&abf[r][c4 * 4] = p;
    }
    __syncthreads();

    const int wave = tid >> 5, lane = tid & 31;
    const int m = lane & 15, hi = lane >> 4;
    const int col = wave * 16 + (lane & 15);
    const int kb0 = 8 * hi;

    FragC acc0, acc1;
#pragma unroll
    for (int i = 0; i < 8; ++i) { acc0.f[i] = 0.f; acc1.f[i] = 0.f; }

    const unsigned short* brow = Wbf + (size_t)col * FDIM;
#pragma unroll
    for (int kc = 0; kc < 4; kc += 2) {
        FragAB a0 = frag_from_u16(&abf[m][0], kc * 32 + kb0);
        FragAB b0 = frag_from_u16(brow,        kc * 32 + kb0);
        acc0.v = __builtin_amdgcn_wmma_f32_16x16x32_bf16(
            false, a0.v, false, b0.v, (short)0, acc0.v, false, false);
        FragAB a1 = frag_from_u16(&abf[m][0], (kc + 1) * 32 + kb0);
        FragAB b1 = frag_from_u16(brow,        (kc + 1) * 32 + kb0);
        acc1.v = __builtin_amdgcn_wmma_f32_16x16x32_bf16(
            false, a1.v, false, b1.v, (short)0, acc1.v, false, false);
    }
#pragma unroll
    for (int v = 0; v < 8; ++v) {
        const int row = v + 8 * hi;                         // C layout: M = v + 8*hi
        node_proj[(size_t)(blockIdx.x * 16 + row) * FDIM + col] = acc0.f[v] + acc1.f[v];
    }
}

// ---------------------------------------------------------------------------
// Kernel 3 (hot loop): one block per node (32768 blocks).
//   nap = neighbors@B1^T + aspects@B2^T + ba
//   out = elu( sum_k exp(leakyrelu(nap + node_proj)) * attn + bias )
// 16x16 C tile = all 16 edges x 16 cols -> k-reduction stays in-fragment.
// Neighbor and aspect GEMMs run as two interleaved accumulator chains.
// ---------------------------------------------------------------------------
__global__ __launch_bounds__(256)
void gat_edge_kernel(const float* __restrict__ neighbors,
                     const float* __restrict__ aspects,
                     const float* __restrict__ attn,
                     const unsigned short* __restrict__ B1bf,
                     const unsigned short* __restrict__ B2bf,
                     const float* __restrict__ ba,
                     const float* __restrict__ bias,
                     const float* __restrict__ node_proj,
                     float* __restrict__ out) {
    __shared__ __align__(16) float          stage[2][KNB * FDIM];  // 16 KB fp32
    __shared__ __align__(16) unsigned short nbbf[KNB][BSTR];       // bf16 tiles
    __shared__ __align__(16) unsigned short asbf[KNB][BSTR];
    __shared__ float at[KNB];

    const int tid  = threadIdx.x;
    const int node = blockIdx.x;
    const size_t base = (size_t)node * KNB * FDIM;
    const float* gn = neighbors + base;
    const float* ga = aspects + base;
    const unsigned s0 = (unsigned)(uintptr_t)&stage[0][0];
    const unsigned s1 = (unsigned)(uintptr_t)&stage[1][0];

#pragma unroll
    for (int i = 0; i < 2; ++i) {
        const unsigned off = (unsigned)(tid + i * 256) * 16u;      // 512 x b128 each
        async_copy_b128(s0 + off, gn, off);
        async_copy_b128(s1 + off, ga, off);
    }
    if (tid < KNB) at[tid] = attn[node * KNB + tid];
    wait_async_all();
    __syncthreads();

    // fp32 -> bf16 (padded tiles) : each thread converts 8 floats per tile
#pragma unroll
    for (int i = 0; i < 2; ++i) {
        const int t  = tid + i * 256;
        const int r  = t >> 5;
        const int c4 = t & 31;
        const float4 v0 = ((const float4*)stage[0])[t];
        uint2 p0; p0.x = pack_bf16x2(v0.x, v0.y); p0.y = pack_bf16x2(v0.z, v0.w);
        *(uint2*)&nbbf[r][c4 * 4] = p0;
        const float4 v1 = ((const float4*)stage[1])[t];
        uint2 p1; p1.x = pack_bf16x2(v1.x, v1.y); p1.y = pack_bf16x2(v1.z, v1.w);
        *(uint2*)&asbf[r][c4 * 4] = p1;
    }
    __syncthreads();

    const int wave = tid >> 5, lane = tid & 31;
    const int m = lane & 15, hi = lane >> 4;
    const int col = wave * 16 + (lane & 15);
    const int kb0 = 8 * hi;

    FragC accN, accA;   // independent chains: neighbors x B1, aspects x B2
#pragma unroll
    for (int i = 0; i < 8; ++i) { accN.f[i] = 0.f; accA.f[i] = 0.f; }

    const unsigned short* b1row = B1bf + (size_t)col * FDIM;
    const unsigned short* b2row = B2bf + (size_t)col * FDIM;

#pragma unroll
    for (int kc = 0; kc < 4; ++kc) {
        const int kb = kc * 32 + kb0;
        FragAB aN = frag_from_u16(&nbbf[m][0], kb);
        FragAB bN = frag_from_u16(b1row,       kb);
        accN.v = __builtin_amdgcn_wmma_f32_16x16x32_bf16(
            false, aN.v, false, bN.v, (short)0, accN.v, false, false);
        FragAB aA = frag_from_u16(&asbf[m][0], kb);
        FragAB bA = frag_from_u16(b2row,       kb);
        accA.v = __builtin_amdgcn_wmma_f32_16x16x32_bf16(
            false, aA.v, false, bA.v, (short)0, accA.v, false, false);
    }

    // Epilogue: lane holds col `col`, rows (edges) v + 8*hi.
    const float nproj = node_proj[(size_t)node * FDIM + col];
    const float bav = ba[col];
    float sum = 0.f;
#pragma unroll
    for (int v = 0; v < 8; ++v) {
        float x = accN.f[v] + accA.f[v] + bav + nproj;
        x = (x > 0.f) ? x : 0.2f * x;                  // LeakyReLU(0.2)
        sum = __builtin_fmaf(__expf(x), at[v + 8 * hi], sum);
    }
    sum += __shfl_xor(sum, 16, 32);                    // fold the two lane halves
    if (hi == 0) {
        float y = sum + bias[col];
        y = (y > 0.f) ? y : (__expf(y) - 1.f);         // ELU
        out[(size_t)node * FDIM + col] = y;
    }
}

// ---------------------------------------------------------------------------
// Host-side launcher
// ---------------------------------------------------------------------------
extern "C" void kernel_launch(void* const* d_in, const int* in_sizes, int n_in,
                              void* d_out, int out_size, void* d_ws, size_t ws_size,
                              hipStream_t stream) {
    const float* nodes     = (const float*)d_in[0];
    const float* neighbors = (const float*)d_in[1];
    const float* aspects   = (const float*)d_in[2];
    const float* attn      = (const float*)d_in[3];
    const float* W         = (const float*)d_in[4];
    const float* Wa        = (const float*)d_in[5];
    const float* ba        = (const float*)d_in[6];
    const float* bias      = (const float*)d_in[7];
    float* out = (float*)d_out;

    const int N = in_sizes[0] / FDIM;   // 32768

    // Workspace layout
    char* ws = (char*)d_ws;
    unsigned short* B1bf = (unsigned short*)(ws);                    // 32 KB
    unsigned short* B2bf = (unsigned short*)(ws + 32768);            // 32 KB
    unsigned short* Wbf  = (unsigned short*)(ws + 65536);            // 32 KB
    float*          nprj = (float*)(ws + 98304);                     // N*128*4

    prep_weights_kernel<<<FDIM, FDIM, 0, stream>>>(W, Wa, B1bf, B2bf, Wbf);
    node_proj_kernel<<<N / 16, 256, 0, stream>>>(nodes, Wbf, nprj);
    gat_edge_kernel<<<N, 256, 0, stream>>>(neighbors, aspects, attn,
                                           B1bf, B2bf, ba, bias, nprj, out);
}
